// Transformer_45028437131955
// MI455X (gfx1250) — compile-verified, matrix operand order checked by reading, not run on hardware
//
#include <hip/hip_runtime.h>
#include <hip/hip_bf16.h>

// ---------------------------------------------------------------------------
// Transformer forward for MI455X (gfx1250, wave32, WMMA).
// All GEMMs via v_wmma_f32_16x16x32_bf16 (bf16 in, f32 accumulate).
// 256-thread blocks, 128x128 tile, wave tile 32x64 (8 WMMA / wave / K-step),
// double-buffered LDS, float4 global loads, native v_cvt f32->bf16 staging,
// global_prefetch for the K+2 tile, templated on transB/ReLU/residual.
// ---------------------------------------------------------------------------

typedef __attribute__((ext_vector_type(16))) __bf16 v16bf;
typedef __attribute__((ext_vector_type(8)))  __bf16 v8bf;
typedef __attribute__((ext_vector_type(4)))  __bf16 v4bf;
typedef __attribute__((ext_vector_type(8)))  float  v8f;

union FragBF { v16bf f; v8bf h[2]; };

// native hardware convert (v_cvt_*bf16*), RNE — much cheaper than bit math
static __device__ __forceinline__ __bf16 f2bf(float f) { return (__bf16)f; }

#define BM 128
#define BN 128
#define BK 32

// C[M,N] = act( alpha * A[M,K] * B[K,N] + bias + resid )
// TRANSB: B stored [N,K] in memory. RELU: clamp at 0. RESID: C += resid.
// Batched over blockIdx.z with element strides sA/sB/sC.
// Requires M%128==0, K%32==0, lda/ldb multiples of 4; N%128==0 unless
// TRANSB==0 (then clamped loads + guarded epilogue handle N%4==0).
template<int TRANSB, int RELU, int RESID>
__global__ __launch_bounds__(256)
void gemm_wmma(const float* __restrict__ A, int lda, long long sA,
               const float* __restrict__ Bm, int ldb, long long sB,
               const float* __restrict__ bias,
               const float* __restrict__ resid,
               float* __restrict__ C, int ldc, long long sC,
               int N, int K, float alpha)
{
    __shared__ alignas(16) __bf16 As[2][BM][BK];   // [m][k]
    __shared__ alignas(16) __bf16 Bs[2][BN][BK];   // [n][k] (k-contiguous)

    const int tid = threadIdx.x;
    const int z   = blockIdx.z;
    const int m0  = blockIdx.y * BM;
    const int n0  = blockIdx.x * BN;

    A  += (long long)z * sA;
    Bm += (long long)z * sB;
    C  += (long long)z * sC;
    const float* Rz = RESID ? (resid + (long long)z * sC) : nullptr;

    const int lane  = tid & 31;
    const int wv    = tid >> 5;          // 8 waves: wm = wv&3 (M), wn = wv>>2 (N)
    const int wm    = wv & 3;
    const int wn    = wv >> 2;
    const int nc    = lane & 15;
    const int hi    = lane >> 4;
    const int kbase = hi * 8;            // ISA 16-bit striping: K {kb..kb+7, kb+16..kb+23}

    v8f acc[2][4];
    #pragma unroll
    for (int fm = 0; fm < 2; ++fm)
        #pragma unroll
        for (int fn = 0; fn < 4; ++fn)
            #pragma unroll
            for (int e = 0; e < 8; ++e) acc[fm][fn][e] = 0.0f;

    float4 ra[4], rb[4];

    auto loadA = [&](int k0) {
        #pragma unroll
        for (int it = 0; it < 4; ++it) {
            const int v = (tid + 256 * it) * 4;       // 0..4095
            const int r = v >> 5, c = v & 31;
            ra[it] = *(const float4*)&A[(size_t)(m0 + r) * lda + (k0 + c)];
        }
    };
    auto loadB = [&](int k0) {
        #pragma unroll
        for (int it = 0; it < 4; ++it) {
            const int v = (tid + 256 * it) * 4;
            if (TRANSB) {                              // Bmem[N][K], k contiguous
                const int r = v >> 5, c = v & 31;
                rb[it] = *(const float4*)&Bm[(size_t)(n0 + r) * ldb + (k0 + c)];
            } else {                                   // Bmem[K][N], n contiguous
                const int kk = v >> 7, n = v & 127;
                int col = n0 + n;
                if (col > N - 4) col = N - 4;          // clamp (N%4==0)
                rb[it] = *(const float4*)&Bm[(size_t)(k0 + kk) * ldb + col];
            }
        }
    };
    auto stageA = [&](int p) {
        #pragma unroll
        for (int it = 0; it < 4; ++it) {
            const int v = (tid + 256 * it) * 4;
            const int r = v >> 5, c = v & 31;
            v4bf t;
            t[0] = f2bf(ra[it].x); t[1] = f2bf(ra[it].y);
            t[2] = f2bf(ra[it].z); t[3] = f2bf(ra[it].w);
            *(v4bf*)&As[p][r][c] = t;
        }
    };
    auto stageB = [&](int p) {
        #pragma unroll
        for (int it = 0; it < 4; ++it) {
            const int v = (tid + 256 * it) * 4;
            if (TRANSB) {
                const int r = v >> 5, c = v & 31;
                v4bf t;
                t[0] = f2bf(rb[it].x); t[1] = f2bf(rb[it].y);
                t[2] = f2bf(rb[it].z); t[3] = f2bf(rb[it].w);
                *(v4bf*)&Bs[p][r][c] = t;
            } else {
                const int kk = v >> 7, n = v & 127;
                Bs[p][n + 0][kk] = f2bf(rb[it].x);
                Bs[p][n + 1][kk] = f2bf(rb[it].y);
                Bs[p][n + 2][kk] = f2bf(rb[it].z);
                Bs[p][n + 3][kk] = f2bf(rb[it].w);
            }
        }
    };

    // prologue: stage tile 0 into buffer 0
    loadA(0); loadB(0);
    stageA(0); stageB(0);

    int p = 0;
    for (int k0 = 0; k0 < K; k0 += BK) {
        __syncthreads();                 // buffer p ready for all waves
        const bool more = (k0 + BK) < K;
        if (more) { loadA(k0 + BK); loadB(k0 + BK); }   // prefetch next tile -> regs

        // warm L2/WGP$ for the tile after next (global_prefetch_b8)
        if (k0 + 2 * BK < K) {
            const int r = (tid * 4) >> 5, c = (tid * 4) & 31;
            __builtin_prefetch(&A[(size_t)(m0 + r) * lda + (k0 + 2 * BK + c)], 0, 3);
            if (TRANSB)
                __builtin_prefetch(&Bm[(size_t)(n0 + r) * ldb + (k0 + 2 * BK + c)], 0, 3);
            else
                __builtin_prefetch(&Bm[(size_t)(k0 + 2 * BK + (tid >> 5)) * ldb + n0 + ((tid & 31) * 4)], 0, 3);
        }

        FragBF af[2];
        #pragma unroll
        for (int fm = 0; fm < 2; ++fm) {
            const __bf16* ap = &As[p][32 * wm + 16 * fm + nc][0];
            af[fm].h[0] = *(const v8bf*)(ap + kbase);
            af[fm].h[1] = *(const v8bf*)(ap + kbase + 16);
        }
        #pragma unroll
        for (int fn = 0; fn < 4; ++fn) {
            FragBF bf;
            const __bf16* bp = &Bs[p][64 * wn + 16 * fn + nc][0];
            bf.h[0] = *(const v8bf*)(bp + kbase);
            bf.h[1] = *(const v8bf*)(bp + kbase + 16);
            acc[0][fn] = __builtin_amdgcn_wmma_f32_16x16x32_bf16(
                false, af[0].f, false, bf.f, (short)0, acc[0][fn], false, false);
            acc[1][fn] = __builtin_amdgcn_wmma_f32_16x16x32_bf16(
                false, af[1].f, false, bf.f, (short)0, acc[1][fn], false, false);
        }
        if (more) { stageA(p ^ 1); stageB(p ^ 1); }
        p ^= 1;
    }

    // epilogue: C/D layout -> lane (nc) = N, VGPR r -> M = r + 8*hi
    #pragma unroll
    for (int fn = 0; fn < 4; ++fn) {
        const int n_g = n0 + 64 * wn + 16 * fn + nc;
        if (n_g < N) {
            const float bz = bias ? bias[n_g] : 0.0f;
            #pragma unroll
            for (int fm = 0; fm < 2; ++fm) {
                #pragma unroll
                for (int r = 0; r < 8; ++r) {
                    const int m_g = m0 + 32 * wm + 16 * fm + r + 8 * hi;
                    float v = alpha * acc[fm][fn][r] + bz;
                    const size_t idx = (size_t)m_g * ldc + n_g;
                    if (RESID) v += Rz[idx];
                    if (RELU) v = fmaxf(v, 0.0f);
                    C[idx] = v;
                }
            }
        }
    }
}

// token embedding * sqrt(D) + sinusoidal positional encoding (on the fly)
// grid: T blocks (t = s*B + b), 128 threads, 4 cols each. D=512, B=4.
__global__ __launch_bounds__(128)
void embed_k(const int* __restrict__ tok, const float* __restrict__ emb,
             float* __restrict__ o)
{
    const int t = blockIdx.x, tid = threadIdx.x;
    const int token = tok[t];
    const float s = (float)(t >> 2);           // B = 4
    const float* e = emb + (size_t)token * 512;
    float* orow = o + (size_t)t * 512;
    #pragma unroll
    for (int k = 0; k < 4; ++k) {
        const int c = tid + 128 * k;
        const float ang = s * __expf(-(float)(c & ~1) * 0.0179889460f); // ln(1e4)/512
        const float pe  = (c & 1) ? __cosf(ang) : __sinf(ang);
        orow[c] = e[c] * 22.627416997969522f + pe;   // sqrt(512)
    }
}

// LayerNorm over last dim (512). One block of 128 per row.
__global__ __launch_bounds__(128)
void layernorm_k(const float* __restrict__ x, const float* __restrict__ w,
                 const float* __restrict__ b, float* __restrict__ o)
{
    __shared__ float red[128];
    const int row = blockIdx.x, tid = threadIdx.x;
    const float* xr = x + (size_t)row * 512;
    float v[4]; float s = 0.0f;
    #pragma unroll
    for (int k = 0; k < 4; ++k) { v[k] = xr[tid + 128 * k]; s += v[k]; }
    red[tid] = s; __syncthreads();
    for (int off = 64; off > 0; off >>= 1) {
        if (tid < off) red[tid] += red[tid + off];
        __syncthreads();
    }
    const float mean = red[0] * (1.0f / 512.0f);
    __syncthreads();
    s = 0.0f;
    #pragma unroll
    for (int k = 0; k < 4; ++k) { const float d = v[k] - mean; s += d * d; }
    red[tid] = s; __syncthreads();
    for (int off = 64; off > 0; off >>= 1) {
        if (tid < off) red[tid] += red[tid + off];
        __syncthreads();
    }
    const float rstd = rsqrtf(red[0] * (1.0f / 512.0f) + 1e-5f);
    float* orow = o + (size_t)row * 512;
    #pragma unroll
    for (int k = 0; k < 4; ++k) {
        const int c = tid + 128 * k;
        orow[c] = (v[k] - mean) * rstd * w[c] + b[c];
    }
}

// row softmax over 512 cols with optional causal mask (col > row blocked).
// grid: (rows=512, batch=B*H), 128 threads.
__global__ __launch_bounds__(128)
void softmax_k(float* __restrict__ sc, int causal)
{
    __shared__ float red[128];
    const int q = blockIdx.x, z = blockIdx.y, tid = threadIdx.x;
    float* row = sc + ((size_t)z * 512 + q) * 512;
    float v[4]; float mx = -3.0e38f;
    #pragma unroll
    for (int k = 0; k < 4; ++k) {
        const int c = tid + 128 * k;
        float t = row[c];
        if (causal && c > q) t = -3.0e38f;
        v[k] = t; mx = fmaxf(mx, t);
    }
    red[tid] = mx; __syncthreads();
    for (int off = 64; off > 0; off >>= 1) {
        if (tid < off) red[tid] = fmaxf(red[tid], red[tid + off]);
        __syncthreads();
    }
    mx = red[0]; __syncthreads();
    float s = 0.0f;
    #pragma unroll
    for (int k = 0; k < 4; ++k) { v[k] = __expf(v[k] - mx); s += v[k]; }
    red[tid] = s; __syncthreads();
    for (int off = 64; off > 0; off >>= 1) {
        if (tid < off) red[tid] += red[tid + off];
        __syncthreads();
    }
    const float inv = 1.0f / red[0];
    #pragma unroll
    for (int k = 0; k < 4; ++k) row[tid + 128 * k] = v[k] * inv;
}

// ---------------------------------------------------------------------------
extern "C" void kernel_launch(void* const* d_in, const int* in_sizes, int n_in,
                              void* d_out, int out_size, void* d_ws, size_t ws_size,
                              hipStream_t stream)
{
    (void)in_sizes; (void)n_in; (void)out_size; (void)ws_size;

    const int* src = (const int*)d_in[0];
    const int* tgt = (const int*)d_in[1];
    // d_in[2] = tgt_mask: causal, handled structurally in softmax_k
    const float* token_emb  = (const float*)d_in[3];
    const float* enc_attn_w = (const float*)d_in[4];
    const float* enc_attn_b = (const float*)d_in[5];
    const float* enc_ln_w   = (const float*)d_in[6];
    const float* enc_ln_b   = (const float*)d_in[7];
    const float* enc_ffn_w1 = (const float*)d_in[8];
    const float* enc_ffn_b1 = (const float*)d_in[9];
    const float* enc_ffn_w2 = (const float*)d_in[10];
    const float* enc_ffn_b2 = (const float*)d_in[11];
    const float* dec_attn_w = (const float*)d_in[12];
    const float* dec_attn_b = (const float*)d_in[13];
    const float* dec_ln_w   = (const float*)d_in[14];
    const float* dec_ln_b   = (const float*)d_in[15];
    const float* dec_ffn_w1 = (const float*)d_in[16];
    const float* dec_ffn_b1 = (const float*)d_in[17];
    const float* dec_ffn_w2 = (const float*)d_in[18];
    const float* dec_ffn_b2 = (const float*)d_in[19];
    const float* out_w      = (const float*)d_in[20];
    const float* out_b      = (const float*)d_in[21];
    float* out = (float*)d_out;

    const int D = 512, F = 2048, V = 32000, S = 512, T = 2048, BH = 32;
    const long long DD = (long long)D * D;
    const long long TD = (long long)T * D;

    // workspace layout (fp32): x | mem | nx | q k v ctx | scores
    float* xd     = (float*)d_ws;            // decoder residual stream  (T*D)
    float* memb   = xd + TD;                 // encoder stream == memory (T*D)
    float* nx     = memb + TD;               // LN output                (T*D)
    float* qb     = nx + TD;                 // (T*D)
    float* kb     = qb + TD;                 // (T*D)
    float* vb     = kb + TD;                 // (T*D)
    float* ctxb   = vb + TD;                 // (T*D)
    float* scores = ctxb + TD;               // B*H*S*S = 8M floats
    float* ffh    = qb;                      // FFN hidden (T*F) aliases q..ctx

    auto gemm = [&](const float* A, int lda, long long sA,
                    const float* Bm, int ldb, long long sB,
                    const float* bias, const float* resid,
                    float* C, int ldc, long long sC,
                    int M, int N, int K, float alpha,
                    bool transB, bool relu, int batch) {
        dim3 g((unsigned)((N + BN - 1) / BN), (unsigned)(M / BM), (unsigned)batch);
        if (transB)
            gemm_wmma<1, 0, 0><<<g, 256, 0, stream>>>(A, lda, sA, Bm, ldb, sB,
                                                      bias, resid, C, ldc, sC, N, K, alpha);
        else if (relu)
            gemm_wmma<0, 1, 0><<<g, 256, 0, stream>>>(A, lda, sA, Bm, ldb, sB,
                                                      bias, resid, C, ldc, sC, N, K, alpha);
        else if (resid)
            gemm_wmma<0, 0, 1><<<g, 256, 0, stream>>>(A, lda, sA, Bm, ldb, sB,
                                                      bias, resid, C, ldc, sC, N, K, alpha);
        else
            gemm_wmma<0, 0, 0><<<g, 256, 0, stream>>>(A, lda, sA, Bm, ldb, sB,
                                                      bias, resid, C, ldc, sC, N, K, alpha);
    };

    auto ln = [&](const float* xin, const float* w, const float* b) {
        layernorm_k<<<T, 128, 0, stream>>>(xin, w, b, nx);
    };

    // Full multi-head attention: q/k/v proj, QK^T/8, softmax, PV, out-proj+residual.
    auto mha = [&](const float* qin, const float* kin, const float* vin,
                   const float* W, const float* bv, int causal, float* xio) {
        gemm(qin, D, 0, W,          D, 0, bv,         nullptr, qb, D, 0, T, D, D, 1.0f, false, false, 1);
        gemm(kin, D, 0, W + DD,     D, 0, bv + D,     nullptr, kb, D, 0, T, D, D, 1.0f, false, false, 1);
        gemm(vin, D, 0, W + 2 * DD, D, 0, bv + 2 * D, nullptr, vb, D, 0, T, D, D, 1.0f, false, false, 1);
        // scores[z=b*H+h][q][k] = (Q . K) / 8 ; per-head base offset = z*64, row stride = B*D
        gemm(qb, 2048, 64, kb, 2048, 64, nullptr, nullptr,
             scores, S, (long long)S * S, S, S, 64, 0.125f, /*transB*/true, false, BH);
        softmax_k<<<dim3(S, BH), 128, 0, stream>>>(scores, causal);
        // ctx = P @ V back into [T, D] layout (per-head base z*64, ldc = B*D)
        gemm(scores, S, (long long)S * S, vb, 2048, 64, nullptr, nullptr,
             ctxb, 2048, 64, S, 64, S, 1.0f, false, false, BH);
        // out projection + residual add
        gemm(ctxb, D, 0, W + 3 * DD, D, 0, bv + 3 * D, xio, xio, D, 0, T, D, D, 1.0f, false, false, 1);
    };

    auto ffn = [&](const float* w1, const float* b1,
                   const float* w2, const float* b2, float* xio) {
        gemm(nx, D, 0, w1, F, 0, b1, nullptr, ffh, F, 0, T, F, D, 1.0f, false, /*relu*/true, 1);
        gemm(ffh, F, 0, w2, D, 0, b2, xio, xio, D, 0, T, D, F, 1.0f, false, false, 1);
    };

    // ---------------- encoder ----------------
    embed_k<<<T, 128, 0, stream>>>(src, token_emb, memb);
    for (int l = 0; l < 6; ++l) {
        const float* W  = enc_attn_w + (long long)l * 4 * DD;
        const float* bv = enc_attn_b + (long long)l * 4 * D;
        const float* lw = enc_ln_w + (long long)l * 2 * D;
        const float* lb = enc_ln_b + (long long)l * 2 * D;
        ln(memb, lw, lb);
        mha(nx, nx, nx, W, bv, /*causal*/0, memb);
        ln(memb, lw + D, lb + D);
        ffn(enc_ffn_w1 + (long long)l * D * F, enc_ffn_b1 + (long long)l * F,
            enc_ffn_w2 + (long long)l * F * D, enc_ffn_b2 + (long long)l * D, memb);
    }

    // ---------------- decoder ----------------
    embed_k<<<T, 128, 0, stream>>>(tgt, token_emb, xd);
    for (int l = 0; l < 6; ++l) {
        const float* W  = dec_attn_w + (long long)l * 8 * DD;
        const float* bv = dec_attn_b + (long long)l * 8 * D;
        const float* lw = dec_ln_w + (long long)l * 3 * D;
        const float* lb = dec_ln_b + (long long)l * 3 * D;
        ln(xd, lw, lb);
        mha(nx, nx, nx, W, bv, /*causal*/1, xd);                      // self-attn
        ln(xd, lw + D, lb + D);
        mha(nx, memb, memb, W + 4 * DD, bv + 4 * D, /*causal*/0, xd); // cross-attn
        ln(xd, lw + 2 * D, lb + 2 * D);
        ffn(dec_ffn_w1 + (long long)l * D * F, dec_ffn_b1 + (long long)l * F,
            dec_ffn_w2 + (long long)l * F * D, dec_ffn_b2 + (long long)l * D, xd);
    }

    // ---------------- output logits: [T, V] ----------------
    gemm(xd, D, 0, out_w, V, 0, out_b, nullptr, out, V, 0, T, V, D, 1.0f, false, false, 1);
}
